// GaussianMLPBasisLayer_13838384628108
// MI455X (gfx1250) — matrix operand order repaired and checked by reading, not compiled, not run on hardware
//
#include <hip/hip_runtime.h>
#include <hip/hip_bf16.h>
#include <math.h>

// ---------------------------------------------------------------------------
// GaussianMLPBasisLayer for MI455X (gfx1250, wave32, WMMA bf16 16x16x32)
// ---------------------------------------------------------------------------

typedef __bf16 bf16_t;
typedef __attribute__((ext_vector_type(16))) __bf16 v16bf;
typedef __attribute__((ext_vector_type(8)))  float  v8f;

#define NT 32768      // total instances (B*N)
#define DD 512        // feature dim
#define HH 256        // hidden
#define KK 64         // gaussian components
#define SLICES 16     // split-K slices for the statistics pass
#define SLICE_N (NT / SLICES)

static __device__ inline v8f v8f_zero() {
  v8f z;
#pragma unroll
  for (int i = 0; i < 8; ++i) z[i] = 0.0f;
  return z;
}

// ---------------------------------------------------------------------------
// Prepack W1 (512x256 f32) into bf16 B-fragment order:
//   frag f = kc*16 + nt ; within frag: lane(32) x e(16)
//   B element: K = kc*32 + (lane/16)*8 + (e<8 ? e : e+8) ; N = nt*16 + (lane%16)
// ---------------------------------------------------------------------------
__global__ __launch_bounds__(256) void k_prepack_w1(const float* __restrict__ W1,
                                                    bf16_t* __restrict__ pk) {
  int idx = blockIdx.x * 256 + threadIdx.x;          // < 131072
  int f = idx >> 9, r = idx & 511;
  int lane = r >> 4, e = r & 15;
  int kc = f >> 4, nt = f & 15;
  int half = lane >> 4, lnN = lane & 15;
  int K = kc * 32 + half * 8 + (e < 8 ? e : e + 8);
  int N = nt * 16 + lnN;
  pk[idx] = (bf16_t)W1[(size_t)K * HH + N];
}

// ---------------------------------------------------------------------------
// Prepack W2^T (64x256) into bf16 A-fragment order:
//   frag f = ct*8 + kc2 ; A element: m = ct*16 + lane%16,
//   K = kc2*32 + (lane/16)*16 + e
// ---------------------------------------------------------------------------
__global__ __launch_bounds__(256) void k_prepack_w2t(const float* __restrict__ W2,
                                                     bf16_t* __restrict__ pk) {
  int idx = blockIdx.x * 256 + threadIdx.x;          // < 16384
  int f = idx >> 9, r = idx & 511;
  int lane = r >> 4, e = r & 15;
  int ct = f >> 3, kc2 = f & 7;
  int half = lane >> 4, lnN = lane & 15;
  int K = kc2 * 32 + half * 16 + e;
  int m = ct * 16 + lnN;
  pk[idx] = (bf16_t)W2[(size_t)K * KK + m];
}

// ---------------------------------------------------------------------------
// Pass 1: attention MLP. Each wave handles a 16-instance row tile.
//   h = softplus(x@W1 + b1)          (16 kc x 16 nt WMMAs, f32 acc)
//   logits^T = W2^T @ h^T            (8 kc2 x 4 ct WMMAs; h staged in LDS)
//   att = softmax(logits + b2)       (lane <-> lane^16 exchange)
//   store att^T bf16 [64][NT]
// ---------------------------------------------------------------------------
__global__ __launch_bounds__(256) void k_attention(
    const float* __restrict__ x, const float* __restrict__ b1v,
    const float* __restrict__ b2v, const bf16_t* __restrict__ pkW1,
    const bf16_t* __restrict__ pkW2T, bf16_t* __restrict__ attT) {
  alignas(32) __shared__ bf16_t hl_all[8 * 16 * HH];  // 64 KB: per-wave h tile
  const int tid = threadIdx.x;
  const int wave = tid >> 5, lane = tid & 31;
  const int half = lane >> 4, ln = lane & 15;
  const int tile = blockIdx.x * 8 + wave;            // 0..2047
  const int rowBase = tile * 16;
  bf16_t* hl = hl_all + wave * (16 * HH);            // row-major [16][256]

  // ---- GEMM1: x @ W1 -> h (f32 acc) ----
  v8f acc[16];
#pragma unroll
  for (int i = 0; i < 16; ++i) acc[i] = v8f_zero();

  for (int kc = 0; kc < 16; ++kc) {
    const float* p0 = x + (size_t)(rowBase + ln) * DD + kc * 32 + half * 8;
    v16bf a;
#pragma unroll
    for (int e = 0; e < 8; ++e) a[e] = (bf16_t)p0[e];
#pragma unroll
    for (int e = 0; e < 8; ++e) a[8 + e] = (bf16_t)p0[16 + e];
#pragma unroll
    for (int nt = 0; nt < 16; ++nt) {
      v16bf b = *(const v16bf*)(pkW1 + ((size_t)(kc * 16 + nt) * 32 + lane) * 16);
      acc[nt] = __builtin_amdgcn_wmma_f32_16x16x32_bf16(
          false, a, false, b, (short)0, acc[nt], false, false);
    }
  }

  // ---- bias + softplus, stage h (bf16) in LDS row-major [inst16][hid256] ----
#pragma unroll
  for (int nt = 0; nt < 16; ++nt) {
    const int hid = nt * 16 + ln;
    const float bb = b1v[hid];
#pragma unroll
    for (int r = 0; r < 8; ++r) {
      float z = acc[nt][r] + bb;
      float sp = fmaxf(z, 0.0f) + log1pf(expf(-fabsf(z)));   // softplus
      hl[(r + half * 8) * HH + hid] = (bf16_t)sp;
    }
  }
  __syncthreads();

  // ---- GEMM2 (swapped): logits^T = W2^T @ h^T ----
  v8f lg[4];
#pragma unroll
  for (int i = 0; i < 4; ++i) lg[i] = v8f_zero();
  for (int kc2 = 0; kc2 < 8; ++kc2) {
    // B fragment: K = hid = kc2*32 + half*16 + e (contiguous), N = inst = ln
    v16bf bh = *(const v16bf*)(hl + ln * HH + kc2 * 32 + half * 16);
#pragma unroll
    for (int ct = 0; ct < 4; ++ct) {
      v16bf aw = *(const v16bf*)(pkW2T + ((size_t)(ct * 8 + kc2) * 32 + lane) * 16);
      lg[ct] = __builtin_amdgcn_wmma_f32_16x16x32_bf16(
          false, aw, false, bh, (short)0, lg[ct], false, false);
    }
  }

  // ---- softmax over 64 components; lane pair (ln, ln+16) holds all comps ----
  float mx = -3.0e38f;
#pragma unroll
  for (int ct = 0; ct < 4; ++ct)
#pragma unroll
    for (int r = 0; r < 8; ++r) {
      float z = lg[ct][r] + b2v[ct * 16 + r + half * 8];
      lg[ct][r] = z;
      mx = fmaxf(mx, z);
    }
  mx = fmaxf(mx, __shfl_xor(mx, 16, 32));
  float s = 0.0f;
#pragma unroll
  for (int ct = 0; ct < 4; ++ct)
#pragma unroll
    for (int r = 0; r < 8; ++r) {
      float e = expf(lg[ct][r] - mx);
      lg[ct][r] = e;
      s += e;
    }
  s += __shfl_xor(s, 16, 32);
  const float inv = 1.0f / s;
#pragma unroll
  for (int ct = 0; ct < 4; ++ct)
#pragma unroll
    for (int r = 0; r < 8; ++r) {
      int comp = ct * 16 + r + half * 8;
      attT[(size_t)comp * NT + rowBase + ln] = (bf16_t)(lg[ct][r] * inv);
    }
}

// ---------------------------------------------------------------------------
// Pass 2: s1 = att^T @ x, s2 = att^T @ x^2 — deterministic split-K over
// SLICES instance slices. Each wave owns one d-tile and computes all four
// comp-tiles for it (s1 and s2), so the expensive strided x B-fragment is
// loaded/converted once per 8 WMMAs.
// ---------------------------------------------------------------------------
__global__ __launch_bounds__(256) void k_stats(
    const float* __restrict__ x, const bf16_t* __restrict__ attT,
    float* __restrict__ part1, float* __restrict__ part2) {
  const int tid = threadIdx.x, wave = tid >> 5, lane = tid & 31;
  const int half = lane >> 4, ln = lane & 15;
  const int slice = blockIdx.x >> 2;                 // 0..15
  const int tb = blockIdx.x & 3;                     // 0..3
  const int dt = tb * 8 + wave;                      // d tile 0..31

  v8f s1[4], s2[4];
#pragma unroll
  for (int i = 0; i < 4; ++i) { s1[i] = v8f_zero(); s2[i] = v8f_zero(); }

  const int instBase = slice * SLICE_N;

  for (int ch = 0; ch < SLICE_N / 32; ++ch) {
    const int ib = instBase + ch * 32;
    // A fragments from att^T for all 4 comp tiles (contiguous 8-elt runs)
    v16bf a[4];
#pragma unroll
    for (int ct = 0; ct < 4; ++ct) {
      const bf16_t* ap = attT + (size_t)(ct * 16 + ln) * NT + ib + half * 16;
#pragma unroll
      for (int e = 0; e < 8; ++e) a[ct][e] = ap[e];
#pragma unroll
      for (int e = 0; e < 8; ++e) a[ct][8 + e] = ap[8 + e];
    }
    // B fragment from x (strided); square in-register for s2
    const float* xp = x + (size_t)(ib + half * 16) * DD + dt * 16 + ln;
    __builtin_prefetch(xp + (size_t)32 * DD, 0, 1);  // next chunk
    v16bf b, b2;
#pragma unroll
    for (int e = 0; e < 16; ++e) {
      float v = xp[(size_t)e * DD];
      b[e] = (bf16_t)v;
      b2[e] = (bf16_t)(v * v);
    }
#pragma unroll
    for (int ct = 0; ct < 4; ++ct) {
      s1[ct] = __builtin_amdgcn_wmma_f32_16x16x32_bf16(false, a[ct], false, b,  (short)0, s1[ct], false, false);
      s2[ct] = __builtin_amdgcn_wmma_f32_16x16x32_bf16(false, a[ct], false, b2, (short)0, s2[ct], false, false);
    }
  }

  float* p1 = part1 + (size_t)slice * (KK * DD);
  float* p2 = part2 + (size_t)slice * (KK * DD);
#pragma unroll
  for (int ct = 0; ct < 4; ++ct)
#pragma unroll
    for (int r = 0; r < 8; ++r) {
      int comp = ct * 16 + r + half * 8;
      int d = dt * 16 + ln;
      p1[comp * DD + d] = s1[ct][r];
      p2[comp * DD + d] = s2[ct][r];
    }
}

// ---------------------------------------------------------------------------
// sum_att[k] = sum_n att[n][k]
// ---------------------------------------------------------------------------
__global__ __launch_bounds__(256) void k_sumatt(const bf16_t* __restrict__ attT,
                                                float* __restrict__ sum_att) {
  const int k = blockIdx.x;
  const int tid = threadIdx.x;
  __shared__ float red[256];
  float s = 0.0f;
  for (int i = tid; i < NT; i += 256) s += (float)attT[(size_t)k * NT + i];
  red[tid] = s;
  __syncthreads();
  for (int o = 128; o > 0; o >>= 1) {
    if (tid < o) red[tid] += red[tid + o];
    __syncthreads();
  }
  if (tid == 0) sum_att[k] = red[0];
}

// ---------------------------------------------------------------------------
// Finalize: reduce slices, mean/var (reference formula), write var to output,
// pack inv_var & mean*inv_var as bf16 B-fragments, reduce c1/c2 constants.
// ---------------------------------------------------------------------------
__global__ __launch_bounds__(256) void k_finalize(
    const float* __restrict__ part1, const float* __restrict__ part2,
    const float* __restrict__ sum_att, float* __restrict__ outVar,
    bf16_t* __restrict__ pkiv, bf16_t* __restrict__ pkmiv,
    float* __restrict__ c1, float* __restrict__ c2) {
  const int k = blockIdx.x, tid = threadIdx.x;
  const float sa = sum_att[k];
  __shared__ float r1[256], r2[256];
  float lc1 = 0.0f, lc2 = 0.0f;
  for (int d = tid; d < DD; d += 256) {
    float s1 = 0.0f, s2 = 0.0f;
    for (int s = 0; s < SLICES; ++s) {
      s1 += part1[(size_t)s * (KK * DD) + k * DD + d];
      s2 += part2[(size_t)s * (KK * DD) + k * DD + d];
    }
    float mean = s1 / sa;
    float var = (s2 - 2.0f * mean * s1 + sa * mean * mean) / sa;
    outVar[k * DD + d] = var;
    float iv = 1.0f / var;
    float miv = mean * iv;
    // B-fragment packed position for the log-likelihood GEMM
    int ct = k >> 4, lnN = k & 15;
    int kc = d >> 5, rem = d & 31, hf = (rem >> 3) & 1, e = (rem & 7) + ((rem >> 4) << 3);
    size_t off = ((size_t)(kc * 4 + ct) * 32 + hf * 16 + lnN) * 16 + e;
    pkiv[off] = (bf16_t)iv;
    pkmiv[off] = (bf16_t)miv;
    lc1 += mean * mean * iv;
    lc2 += logf(var);
  }
  r1[tid] = lc1;
  r2[tid] = lc2;
  __syncthreads();
  for (int o = 128; o > 0; o >>= 1) {
    if (tid < o) { r1[tid] += r1[tid + o]; r2[tid] += r2[tid + o]; }
    __syncthreads();
  }
  if (tid == 0) { c1[k] = r1[0]; c2[k] = r2[0]; }
}

// ---------------------------------------------------------------------------
// Pass 3: log-likelihood.
//   quad = x^2 @ inv_var^T - 2 x @ (mean*inv_var)^T + c1[k]
//   out  = -0.5*quad - 0.5*c2[k] - 512*ln(2pi)/2
// ---------------------------------------------------------------------------
__global__ __launch_bounds__(256) void k_loglik(
    const float* __restrict__ x, const bf16_t* __restrict__ pkiv,
    const bf16_t* __restrict__ pkmiv, const float* __restrict__ c1,
    const float* __restrict__ c2, float* __restrict__ out) {
  const int tid = threadIdx.x, wave = tid >> 5, lane = tid & 31;
  const int half = lane >> 4, ln = lane & 15;
  const int tile = blockIdx.x * 8 + wave;
  const int rowBase = tile * 16;

  v8f aq[4], am[4];
#pragma unroll
  for (int i = 0; i < 4; ++i) { aq[i] = v8f_zero(); am[i] = v8f_zero(); }

  for (int kc = 0; kc < 16; ++kc) {
    const float* p0 = x + (size_t)(rowBase + ln) * DD + kc * 32 + half * 8;
    __builtin_prefetch(p0 + 32, 0, 1);
    v16bf a, a2;
#pragma unroll
    for (int e = 0; e < 8; ++e) {
      float v = p0[e];
      a[e] = (bf16_t)v;
      a2[e] = (bf16_t)(v * v);
    }
#pragma unroll
    for (int e = 0; e < 8; ++e) {
      float v = p0[16 + e];
      a[8 + e] = (bf16_t)v;
      a2[8 + e] = (bf16_t)(v * v);
    }
#pragma unroll
    for (int ct = 0; ct < 4; ++ct) {
      v16bf biv = *(const v16bf*)(pkiv + ((size_t)(kc * 4 + ct) * 32 + lane) * 16);
      v16bf bmv = *(const v16bf*)(pkmiv + ((size_t)(kc * 4 + ct) * 32 + lane) * 16);
      aq[ct] = __builtin_amdgcn_wmma_f32_16x16x32_bf16(false, a2, false, biv, (short)0, aq[ct], false, false);
      am[ct] = __builtin_amdgcn_wmma_f32_16x16x32_bf16(false, a, false, bmv, (short)0, am[ct], false, false);
    }
  }

  const float CONST = 470.49652900079f;  // 512 * ln(2*pi)/2
#pragma unroll
  for (int ct = 0; ct < 4; ++ct) {
    const int comp = ct * 16 + ln;
    const float cc1 = c1[comp];
    const float cc2 = c2[comp];
#pragma unroll
    for (int r = 0; r < 8; ++r) {
      int inst = rowBase + r + half * 8;
      float quad = aq[ct][r] - 2.0f * am[ct][r] + cc1;
      out[(size_t)inst * KK + comp] = -0.5f * quad - 0.5f * cc2 - CONST;
    }
  }
}

// ---------------------------------------------------------------------------
// Host launcher
// ---------------------------------------------------------------------------
extern "C" void kernel_launch(void* const* d_in, const int* in_sizes, int n_in,
                              void* d_out, int out_size, void* d_ws, size_t ws_size,
                              hipStream_t stream) {
  (void)in_sizes; (void)n_in; (void)out_size; (void)ws_size;
  const float* x  = (const float*)d_in[0];   // [NT, 512]
  const float* W1 = (const float*)d_in[1];   // [512, 256]
  const float* b1 = (const float*)d_in[2];   // [256]
  const float* W2 = (const float*)d_in[3];   // [256, 64]
  const float* b2 = (const float*)d_in[4];   // [64]

  float* out_log = (float*)d_out;                       // [NT, 64]
  float* out_var = (float*)d_out + (size_t)NT * KK;     // [64, 512]

  char* w = (char*)d_ws;
  bf16_t* attT    = (bf16_t*)(w);                                     // 4 MB
  float*  part1   = (float*)(w + (size_t)(4 << 20));                  // 2 MB
  float*  part2   = (float*)(w + (size_t)(6 << 20));                  // 2 MB
  float*  sum_att = (float*)(w + (size_t)(8 << 20));                  // 256 B
  float*  c1      = (float*)(w + (size_t)(8 << 20) + 1024);
  float*  c2      = (float*)(w + (size_t)(8 << 20) + 2048);
  bf16_t* pkW1    = (bf16_t*)(w + (size_t)(8 << 20) + 4096);          // 256 KB
  bf16_t* pkW2T   = (bf16_t*)(w + (size_t)(8 << 20) + 4096 + 262144); // 32 KB
  bf16_t* pkiv    = (bf16_t*)(w + (size_t)(8 << 20) + 4096 + 262144 + 32768);
  bf16_t* pkmiv   = (bf16_t*)(w + (size_t)(8 << 20) + 4096 + 262144 + 32768 + 65536);

  k_prepack_w1 <<<512, 256, 0, stream>>>(W1, pkW1);
  k_prepack_w2t<<< 64, 256, 0, stream>>>(W2, pkW2T);
  k_attention  <<<256, 256, 0, stream>>>(x, b1, b2, pkW1, pkW2T, attT);
  k_stats      <<< 64, 256, 0, stream>>>(x, attT, part1, part2);
  k_sumatt     <<< 64, 256, 0, stream>>>(attT, sum_att);
  k_finalize   <<< 64, 256, 0, stream>>>(part1, part2, sum_att, out_var,
                                         pkiv, pkmiv, c1, c2);
  k_loglik     <<<256, 256, 0, stream>>>(x, pkiv, pkmiv, c1, c2, out_log);
}
